// LMModel_att_10058813407977
// MI455X (gfx1250) — matrix operand order repaired and checked by reading, not compiled, not run on hardware
//
#include <hip/hip_runtime.h>
#include <hip/hip_bf16.h>

// ---------------------------------------------------------------------------
// CDNA5 (gfx1250, wave32) implementation of the attention LM reference.
// All dense math runs on V_WMMA_F32_16X16X32_BF16 (bf16 in, f32 accumulate).
// Sequential GRU / decode scans are single persistent 512-thread workgroups.
// ---------------------------------------------------------------------------

#define NVOC 10000
#define NIN  1024
#define NHID 256
#define H3   768
#define TT   128
#define BB   64
#define ROWS (TT * BB)          // 8192
#define LOGITS_ELEMS ((long)ROWS * NVOC)

typedef __attribute__((ext_vector_type(16))) __bf16         v16bf;
typedef __attribute__((ext_vector_type(8)))  float          v8f;
typedef __attribute__((ext_vector_type(8)))  unsigned short v8us;

union FragU { v16bf bf; v8us us[2]; };

__device__ __forceinline__ unsigned short f2bf(float f) {
  union { float f; unsigned u; } v; v.f = f;
  unsigned u = v.u;
  u += 0x7FFFu + ((u >> 16) & 1u);          // round-to-nearest-even
  return (unsigned short)(u >> 16);
}

__device__ __forceinline__ float sigm(float x) { return 1.0f / (1.0f + __expf(-x)); }

// A fragment: 16x32 bf16 tile of row-major X[*, lda], rows row0..row0+15, K k0..k0+31.
// A k order used here (contiguous): lanes 0-15 -> M=lane, K 0..15; lanes 16-31 -> K 16..31.
__device__ __forceinline__ v16bf load_a(const unsigned short* X, int lda, int row0, int k0) {
  const int lane  = threadIdx.x & 31;
  const int m     = lane & 15;
  const int khalf = (lane >> 4) & 1;        // 0: K base 0, 1: K base 16
  const unsigned short* p = X + (long)(row0 + m) * lda + k0 + khalf * 16;
  FragU u;
  u.us[0] = *(const v8us*)(p);              // K = kb .. kb+7
  u.us[1] = *(const v8us*)(p + 8);          // K = kb+8 .. kb+15
  return u.bf;
}

// B fragment: 32x16 bf16 tile where B(k,n) = W[n][k], W row-major [N, ldw].
// B k order used here (contiguous, same as load_a): lanes 0-15 -> N=lane, K 0..15; lanes 16-31 -> N=lane-16, K 16..31.
__device__ __forceinline__ v16bf load_b(const unsigned short* W, int ldw, int n0, int k0) {
  const int lane  = threadIdx.x & 31;
  const int n     = lane & 15;
  const int khalf = (lane >> 4) & 1;
  const unsigned short* p = W + (long)(n0 + n) * ldw + k0 + khalf * 16;
  FragU u;
  u.us[0] = *(const v8us*)(p);
  u.us[1] = *(const v8us*)(p + 8);
  return u.bf;
}

// Branchless N-edge handling: clamp the row index (duplicate loads, guarded store).
// Keeps EXEC fully set -> clean WMMA codegen (no exec save/restore, no phi movs).
__device__ __forceinline__ v16bf load_b_clamped(const unsigned short* W, int ldw,
                                                int n0, int k0, int N) {
  const int lane = threadIdx.x & 31;
  int n = n0 + (lane & 15);
  n = n < N ? n : N - 1;
  const int khalf = (lane >> 4) & 1;
  const unsigned short* p = W + (long)n * ldw + k0 + khalf * 16;
  FragU u;
  u.us[0] = *(const v8us*)(p);
  u.us[1] = *(const v8us*)(p + 8);
  return u.bf;
}

#define WMMA_BF16(a, b, c) \
  __builtin_amdgcn_wmma_f32_16x16x32_bf16(false, (a), false, (b), (short)0, (c), false, false)

// Wave-cooperative tile GEMM: Y[Mtiles*16, Ntiles*16] = A @ W^T (+bias), bf16 in f32 out.
// Used inside the persistent scan kernels. K must be a multiple of 64.
__device__ __forceinline__ void wave_gemm_tiles(const unsigned short* A, int lda,
                                                const unsigned short* W, int ldw,
                                                const float* bias,
                                                float* Y, int ldy,
                                                int Mtiles, int Ntiles, int K, int nwaves) {
  const int wv    = threadIdx.x >> 5;
  const int lane  = threadIdx.x & 31;
  const int total = Mtiles * Ntiles;
  for (int tIdx = wv; tIdx < total; tIdx += nwaves) {
    const int tm = tIdx % Mtiles;
    const int tn = tIdx / Mtiles;
    v8f acc;
    for (int g = 0; g < 8; ++g) acc[g] = 0.0f;
    for (int k = 0; k < K; k += 64) {
      // issue all loads for both K-subtiles, then consume (loads return in order)
      v16bf a0 = load_a(A, lda, tm * 16, k);
      v16bf b0 = load_b(W, ldw, tn * 16, k);
      v16bf a1 = load_a(A, lda, tm * 16, k + 32);
      v16bf b1 = load_b(W, ldw, tn * 16, k + 32);
      acc = WMMA_BF16(a0, b0, acc);
      acc = WMMA_BF16(a1, b1, acc);
    }
    const int n  = tn * 16 + (lane & 15);
    const int mh = ((lane >> 4) & 1) * 8;
    const float bv = bias ? bias[n] : 0.0f;
    for (int g = 0; g < 8; ++g)
      Y[(tm * 16 + mh + g) * ldy + n] = acc[g] + bv;
  }
}

// ---------------------------------------------------------------------------
// Batched GEMM: Y[M,N] = A[M,K](bf16) @ W[N,K]^T(bf16) + bias, f32 out.
// Block = 256 threads (8 waves). Each wave owns a 32(M) x 64(N) strip
// (2x4 accumulators: each B frag feeds 2 wmmas, each A frag 4; 8 independent
// wmmas per K-step fill the WMMA->VALU hazard slots).
// grid = (ceil(N/512), M/32). K must be a multiple of 64.
// ---------------------------------------------------------------------------
__global__ void __launch_bounds__(256)
k_gemm_bf16(const unsigned short* __restrict__ A, int lda,
            const unsigned short* __restrict__ W, int ldw,
            const float* __restrict__ bias,
            float* __restrict__ Y, long ldy,
            int N, int K) {
  const int row0 = blockIdx.y * 32;
  const int wv   = threadIdx.x >> 5;
  const int lane = threadIdx.x & 31;
  const int n0   = blockIdx.x * 512 + wv * 64;
  if (n0 >= N) return;                       // wave-uniform early out

  v8f acc[2][4];
  for (int r = 0; r < 2; ++r)
    for (int i = 0; i < 4; ++i)
      for (int g = 0; g < 8; ++g) acc[r][i][g] = 0.0f;

  for (int k = 0; k < K; k += 64) {
    // issue all fragment loads for this K-step up front
    v16bf a00 = load_a(A, lda, row0,      k);
    v16bf a10 = load_a(A, lda, row0 + 16, k);
    v16bf a01 = load_a(A, lda, row0,      k + 32);
    v16bf a11 = load_a(A, lda, row0 + 16, k + 32);
    v16bf b0[4], b1[4];
#pragma unroll
    for (int i = 0; i < 4; ++i) b0[i] = load_b_clamped(W, ldw, n0 + i * 16, k,      N);
#pragma unroll
    for (int i = 0; i < 4; ++i) b1[i] = load_b_clamped(W, ldw, n0 + i * 16, k + 32, N);
#pragma unroll
    for (int i = 0; i < 4; ++i) {
      acc[0][i] = WMMA_BF16(a00, b0[i], acc[0][i]);
      acc[1][i] = WMMA_BF16(a10, b0[i], acc[1][i]);
    }
#pragma unroll
    for (int i = 0; i < 4; ++i) {
      acc[0][i] = WMMA_BF16(a01, b1[i], acc[0][i]);
      acc[1][i] = WMMA_BF16(a11, b1[i], acc[1][i]);
    }
  }

  const int nl = lane & 15;
  const int mh = ((lane >> 4) & 1) * 8;
#pragma unroll
  for (int r = 0; r < 2; ++r) {
#pragma unroll
    for (int i = 0; i < 4; ++i) {
      const int n = n0 + i * 16 + nl;
      if (n < N) {                           // store-side guard only
        const float bv = bias ? bias[n] : 0.0f;
        for (int g = 0; g < 8; ++g)
          Y[(long)(row0 + r * 16 + mh + g) * ldy + n] = acc[r][i][g] + bv;
      }
    }
  }
}

// f32 -> bf16 weight conversion
__global__ void k_cvt(const float* __restrict__ s, unsigned short* __restrict__ d, int n) {
  const int i = blockIdx.x * blockDim.x + threadIdx.x;
  if (i < n) d[i] = f2bf(s[i]);
}

// Embedding gather + bf16 convert: one block per (t,b) row.
__global__ void k_embed(const int* __restrict__ tokens, const float* __restrict__ E,
                        unsigned short* __restrict__ emb_us) {
  const int row = blockIdx.x;               // t*B + b
  const long tok = (long)tokens[row];
  const float* src = E + tok * NIN;
  unsigned short* dst = emb_us + (long)row * NIN;
  for (int i = threadIdx.x; i < NIN; i += blockDim.x) dst[i] = f2bf(src[i]);
}

// ---------------------------------------------------------------------------
// GRU1 sequential scan (persistent single workgroup, 512 threads = 16 waves).
// Per step: hg = h @ Whh1^T + bhh1 (WMMA), then GRU gating.
// ---------------------------------------------------------------------------
__global__ void __launch_bounds__(512)
k_gru1(const float* __restrict__ xg1,          // [T,B,3H] f32
       const unsigned short* __restrict__ Whh1_us,  // [3H,H] bf16
       const float* __restrict__ bhh1,
       float* __restrict__ h,                  // ws [B,H] f32 (h1 stays here)
       unsigned short* __restrict__ h_us,      // ws [B,H] bf16
       float* __restrict__ hg,                 // ws [B,3H] f32
       unsigned short* __restrict__ out1_us,   // ws [T,B,H] bf16
       float* __restrict__ dout_tail) {        // d_out + T*B*NVOC : final hidden
  const int tid = threadIdx.x;
  for (int i = tid; i < BB * NHID; i += 512) { h[i] = 0.0f; h_us[i] = 0; }
  __threadfence(); __syncthreads();

  for (int t = 0; t < TT; ++t) {
    wave_gemm_tiles(h_us, NHID, Whh1_us, NHID, bhh1, hg, H3, BB / 16, H3 / 16, NHID, 16);
    __threadfence(); __syncthreads();

    const float* xg = xg1 + (long)t * BB * H3;
    for (int i = tid; i < BB * NHID; i += 512) {
      const int b = i >> 8, c = i & 255;
      const float hp = h[i];
      const float xr = xg[b * H3 + c], xz = xg[b * H3 + NHID + c], xn = xg[b * H3 + 2 * NHID + c];
      const float gr = hg[b * H3 + c], gz = hg[b * H3 + NHID + c], gn = hg[b * H3 + 2 * NHID + c];
      const float r  = sigm(xr + gr);
      const float z  = sigm(xz + gz);
      const float nn = tanhf(xn + r * gn);
      const float hn = (1.0f - z) * nn + z * hp;
      h[i] = hn;
      h_us[i] = f2bf(hn);
      out1_us[(long)t * BB * NHID + i] = f2bf(hn);
    }
    __threadfence(); __syncthreads();
  }
  for (int i = tid; i < BB * NHID; i += 512) dout_tail[i] = h[i];  // hidden output [1,B,H]
}

// ---------------------------------------------------------------------------
// Decode sequential scan (persistent single workgroup, 512 threads = 16 waves).
// Per step: q-GEMM, tanh-score reduction, prefix softmax, state = h1 * sum(energy),
// two [64,768] WMMA GEMMs, GRU2 gating -> preds (bf16 for logits GEMM).
// ---------------------------------------------------------------------------
__global__ void __launch_bounds__(512)
k_decode(const float* __restrict__ proj,            // [T,B,H] f32
         const float* __restrict__ h1,              // [B,H] f32 (final GRU1 hidden)
         const float* __restrict__ xg2e,            // [T,B,3H] f32 (emb part + bih2)
         const unsigned short* __restrict__ Wi_us,  // [H,H] bf16
         const float* __restrict__ bi,
         const unsigned short* __restrict__ Wih2l_us, // [3H,1280] bf16, left K=0..255 (ldw 1280)
         const unsigned short* __restrict__ Whh2_us,  // [3H,H] bf16
         const float* __restrict__ bhh2,
         const float* __restrict__ w_att,           // [H]
         float* __restrict__ q,                     // ws [B,H]
         float* __restrict__ scores,                // ws [T,B]
         float* __restrict__ coef,                  // ws [B]
         unsigned short* __restrict__ state_us,     // ws [B,H] bf16
         float* __restrict__ hg2a,                  // ws [B,3H]
         float* __restrict__ hg2b,                  // ws [B,3H]
         float* __restrict__ h2,                    // ws [B,H]
         unsigned short* __restrict__ h2_us,        // ws [B,H] bf16
         unsigned short* __restrict__ preds_us) {   // ws [T,B,H] bf16
  const int tid = threadIdx.x;
  for (int i = tid; i < BB * NHID; i += 512) { h2[i] = 0.0f; h2_us[i] = 0; }
  __threadfence(); __syncthreads();

  for (int t = 0; t < TT; ++t) {
    // q = h2 @ Wi^T + bi
    wave_gemm_tiles(h2_us, NHID, Wi_us, NHID, bi, q, NHID, BB / 16, NHID / 16, NHID, 16);
    __threadfence(); __syncthreads();

    // additive attention scores over prefix j <= t
    const int pairs = (t + 1) * BB;
    for (int i = tid; i < pairs; i += 512) {
      const int j = i >> 6, b = i & 63;
      const float* pr = proj + (long)(j * BB + b) * NHID;
      const float* qb = q + b * NHID;
      float s = 0.0f;
      for (int hx = 0; hx < NHID; ++hx) s += tanhf(pr[hx] + qb[hx]) * w_att[hx];
      scores[j * BB + b] = s;
    }
    __threadfence(); __syncthreads();

    // prefix softmax; state = sum_j energy_j * h1 = h1 * (sum_j energy_j)
    if (tid < BB) {
      float m = -1e30f;
      for (int j = 0; j <= t; ++j) m = fmaxf(m, scores[j * BB + tid]);
      float se = 0.0f;
      for (int j = 0; j <= t; ++j) se += __expf(scores[j * BB + tid] - m);
      float tot = 0.0f;
      for (int j = 0; j <= t; ++j) tot += __expf(scores[j * BB + tid] - m) / se;
      coef[tid] = tot;
    }
    __threadfence(); __syncthreads();

    for (int i = tid; i < BB * NHID; i += 512)
      state_us[i] = f2bf(h1[i] * coef[i >> 8]);
    __threadfence(); __syncthreads();

    // xg-state part and recurrent gates
    wave_gemm_tiles(state_us, NHID, Wih2l_us, NHID + NIN, nullptr, hg2a, H3,
                    BB / 16, H3 / 16, NHID, 16);
    wave_gemm_tiles(h2_us, NHID, Whh2_us, NHID, bhh2, hg2b, H3,
                    BB / 16, H3 / 16, NHID, 16);
    __threadfence(); __syncthreads();

    const float* xge = xg2e + (long)t * BB * H3;
    for (int i = tid; i < BB * NHID; i += 512) {
      const int b = i >> 8, c = i & 255;
      const float hp = h2[i];
      const float xr = xge[b * H3 + c]            + hg2a[b * H3 + c];
      const float xz = xge[b * H3 + NHID + c]     + hg2a[b * H3 + NHID + c];
      const float xn = xge[b * H3 + 2 * NHID + c] + hg2a[b * H3 + 2 * NHID + c];
      const float gr = hg2b[b * H3 + c], gz = hg2b[b * H3 + NHID + c], gn = hg2b[b * H3 + 2 * NHID + c];
      const float r  = sigm(xr + gr);
      const float z  = sigm(xz + gz);
      const float nn = tanhf(xn + r * gn);
      const float hn = (1.0f - z) * nn + z * hp;
      h2[i] = hn;
      h2_us[i] = f2bf(hn);
      preds_us[(long)t * BB * NHID + i] = f2bf(hn);
    }
    __threadfence(); __syncthreads();
  }
}

// ---------------------------------------------------------------------------
// Host-side launch sequence (graph-capture safe: kernels only).
// ---------------------------------------------------------------------------
extern "C" void kernel_launch(void* const* d_in, const int* in_sizes, int n_in,
                              void* d_out, int out_size, void* d_ws, size_t ws_size,
                              hipStream_t stream) {
  (void)in_sizes; (void)n_in; (void)out_size; (void)ws_size;

  const int*   tokens = (const int*)  d_in[0];
  const float* E      = (const float*)d_in[1];
  const float* Wih1   = (const float*)d_in[2];
  const float* Whh1   = (const float*)d_in[3];
  const float* bih1   = (const float*)d_in[4];
  const float* bhh1   = (const float*)d_in[5];
  const float* Wih2   = (const float*)d_in[6];
  const float* Whh2   = (const float*)d_in[7];
  const float* bih2   = (const float*)d_in[8];
  const float* bhh2   = (const float*)d_in[9];
  const float* Wh     = (const float*)d_in[10];
  const float* bh     = (const float*)d_in[11];
  const float* Wi     = (const float*)d_in[12];
  const float* bi     = (const float*)d_in[13];
  const float* w_att  = (const float*)d_in[14];
  const float* Wd     = (const float*)d_in[15];
  const float* bd     = (const float*)d_in[16];

  float* logits = (float*)d_out;                       // [T,B,NVOC]
  float* dout_tail = (float*)d_out + LOGITS_ELEMS;     // [1,B,H]

  // ---- workspace carve-up (256B aligned regions, ~95 MB total) ----
  char* p = (char*)d_ws;
  auto alloc = [&](size_t bytes) -> void* {
    void* r = (void*)p;
    p += (bytes + 255) & ~(size_t)255;
    return r;
  };
  unsigned short* emb_us   = (unsigned short*)alloc((size_t)ROWS * NIN * 2);
  unsigned short* Wih1_us  = (unsigned short*)alloc((size_t)H3 * NIN * 2);
  unsigned short* Whh1_us  = (unsigned short*)alloc((size_t)H3 * NHID * 2);
  unsigned short* Wih2_us  = (unsigned short*)alloc((size_t)H3 * (NHID + NIN) * 2);
  unsigned short* Whh2_us  = (unsigned short*)alloc((size_t)H3 * NHID * 2);
  unsigned short* Wh_us    = (unsigned short*)alloc((size_t)NHID * NHID * 2);
  unsigned short* Wi_us    = (unsigned short*)alloc((size_t)NHID * NHID * 2);
  unsigned short* Wd_us    = (unsigned short*)alloc((size_t)NVOC * NHID * 2);
  float*          xg1      = (float*)alloc((size_t)ROWS * H3 * 4);
  float*          xg2e     = (float*)alloc((size_t)ROWS * H3 * 4);
  unsigned short* out1_us  = (unsigned short*)alloc((size_t)ROWS * NHID * 2);
  float*          proj     = (float*)alloc((size_t)ROWS * NHID * 4);
  float*          hg       = (float*)alloc((size_t)BB * H3 * 4);
  float*          h1       = (float*)alloc((size_t)BB * NHID * 4);
  unsigned short* h1_us    = (unsigned short*)alloc((size_t)BB * NHID * 2);
  float*          q        = (float*)alloc((size_t)BB * NHID * 4);
  float*          scores   = (float*)alloc((size_t)TT * BB * 4);
  float*          coef     = (float*)alloc((size_t)BB * 4);
  unsigned short* state_us = (unsigned short*)alloc((size_t)BB * NHID * 2);
  float*          hg2a     = (float*)alloc((size_t)BB * H3 * 4);
  float*          hg2b     = (float*)alloc((size_t)BB * H3 * 4);
  float*          h2       = (float*)alloc((size_t)BB * NHID * 4);
  unsigned short* h2_us    = (unsigned short*)alloc((size_t)BB * NHID * 2);
  unsigned short* preds_us = (unsigned short*)alloc((size_t)ROWS * NHID * 2);

  // 1) weight conversion to bf16
  auto cvt = [&](const float* s, unsigned short* d, int n) {
    k_cvt<<<dim3((n + 255) / 256), dim3(256), 0, stream>>>(s, d, n);
  };
  cvt(Wih1, Wih1_us, H3 * NIN);
  cvt(Whh1, Whh1_us, H3 * NHID);
  cvt(Wih2, Wih2_us, H3 * (NHID + NIN));
  cvt(Whh2, Whh2_us, H3 * NHID);
  cvt(Wh,   Wh_us,   NHID * NHID);
  cvt(Wi,   Wi_us,   NHID * NHID);
  cvt(Wd,   Wd_us,   NVOC * NHID);

  // 2) embedding gather -> bf16
  k_embed<<<dim3(ROWS), dim3(256), 0, stream>>>(tokens, E, emb_us);

  // 3) xg1 = emb @ Wih1^T + bih1           [8192,768] K=1024
  k_gemm_bf16<<<dim3((H3 + 511) / 512, ROWS / 32), dim3(256), 0, stream>>>(
      emb_us, NIN, Wih1_us, NIN, bih1, xg1, H3, H3, NIN);

  // 4) xg2e = emb @ Wih2[:,H:]^T + bih2    [8192,768] K=1024 (ldw = 1280, col offset 256)
  k_gemm_bf16<<<dim3((H3 + 511) / 512, ROWS / 32), dim3(256), 0, stream>>>(
      emb_us, NIN, Wih2_us + NHID, NHID + NIN, bih2, xg2e, H3, H3, NIN);

  // 5) GRU1 sequential scan (also writes final hidden to d_out tail)
  k_gru1<<<dim3(1), dim3(512), 0, stream>>>(xg1, Whh1_us, bhh1, h1, h1_us, hg,
                                            out1_us, dout_tail);

  // 6) proj = out1 @ Wh^T + bh             [8192,256] K=256
  k_gemm_bf16<<<dim3(1, ROWS / 32), dim3(256), 0, stream>>>(
      out1_us, NHID, Wh_us, NHID, bh, proj, NHID, NHID, NHID);

  // 7) decode sequential scan
  k_decode<<<dim3(1), dim3(512), 0, stream>>>(proj, h1, xg2e, Wi_us, bi,
                                              Wih2_us, Whh2_us, bhh2, w_att,
                                              q, scores, coef, state_us,
                                              hg2a, hg2b, h2, h2_us, preds_us);

  // 8) logits = preds @ Wd^T + bd          [8192,10000] K=256 (output-BW bound)
  k_gemm_bf16<<<dim3((NVOC + 511) / 512, ROWS / 32), dim3(256), 0, stream>>>(
      preds_us, NHID, Wd_us, NHID, bd, logits, NVOC, NVOC, NHID);
}
